// HeteroGNN_6691559047207
// MI455X (gfx1250) — matrix operand order repaired and checked by reading, not compiled, not run on hardware
//
#include <hip/hip_runtime.h>

typedef __attribute__((ext_vector_type(16))) __bf16 v16bf;
typedef __attribute__((ext_vector_type(8)))  float  v8f;

#define NDRUG 50000
#define NDIS  50000
#define INCH  256
#define HID   512
#define OUTCH 256
#define NEDGE 500000
#define NLAYERS 3

#define BM 128
#define BN 128
#define LDA 40   // padded LDS row stride (elements): keeps 16B alignment + staggers banks

#if defined(__has_builtin)
#if __has_builtin(__builtin_amdgcn_global_load_async_to_lds_b128)
#define USE_ASYNC 1
#endif
#endif

#if defined(__has_builtin)
#if __has_builtin(__builtin_amdgcn_s_wait_asynccnt)
#define WAIT_ASYNC(n) __builtin_amdgcn_s_wait_asynccnt(n)
#else
#define WAIT_ASYNC(n) asm volatile("s_wait_asynccnt %0" ::"n"(n) : "memory")
#endif
#else
#define WAIT_ASYNC(n)
#endif

// ---------------------------------------------------------------------------
// helpers
// ---------------------------------------------------------------------------
__device__ __forceinline__ void load_frag16(v16bf* f, const __bf16* p0, const __bf16* p1) {
  uint4* q = reinterpret_cast<uint4*>(f);
  q[0] = *reinterpret_cast<const uint4*>(p0);
  q[1] = *reinterpret_cast<const uint4*>(p1);
}

#ifdef USE_ASYNC
// builtin prototype: (int4 AS1*, int4 AS3*, imm offset, imm cpol) with GCC-vector int4
typedef int async_i4 __attribute__((vector_size(16)));
typedef __attribute__((address_space(1))) async_i4* as1_i4p;
typedef __attribute__((address_space(3))) async_i4* as3_i4p;

// async copy of 16 bytes (8 bf16) global -> LDS, tracked by ASYNCcnt
__device__ __forceinline__ void async_cp_b128(const __bf16* g, __bf16* l) {
  __builtin_amdgcn_global_load_async_to_lds_b128((as1_i4p)g, (as3_i4p)l, 0, 0);
}
#endif

// One 32-wide k-slab: load A fragments from LDS tile + B fragments from packed
// global weights, then 8 WMMAs.
__device__ __forceinline__ void compute_slab(const __bf16* buf,
                                             const __bf16* __restrict__ Bp,
                                             int k0, int N, int n0w,
                                             int wm, int lane, v8f (&acc)[2][4]) {
  const int ml = lane & 15;
  const int kh = lane >> 4;
  v16bf af[2];
#pragma unroll
  for (int i = 0; i < 2; ++i) {
    const __bf16* base = buf + (wm * 32 + i * 16 + ml) * LDA + kh * 16;
    load_frag16(&af[i], base, base + 8);
  }
  const __bf16* bbase =
      Bp + ((size_t)(k0 >> 5) * (N >> 4) + (n0w >> 4)) * 512 + lane * 16;
  v16bf bfrag[4];
#pragma unroll
  for (int j = 0; j < 4; ++j)
    load_frag16(&bfrag[j], bbase + (size_t)j * 512, bbase + (size_t)j * 512 + 8);
#pragma unroll
  for (int i = 0; i < 2; ++i)
#pragma unroll
    for (int j = 0; j < 4; ++j)
      acc[i][j] = __builtin_amdgcn_wmma_f32_16x16x32_bf16(
          false, af[i], false, bfrag[j], (short)0, acc[i][j], false, false);
}

// ---------------------------------------------------------------------------
// K-loop for one A/B pair
// ---------------------------------------------------------------------------
#ifdef USE_ASYNC

__device__ __forceinline__ void issue_tile(const __bf16* __restrict__ A, int K, int M,
                                           int grow, int r, int half, int k0,
                                           __bf16* buf) {
  if (k0 < K && grow < M) {
    const __bf16* g = A + (size_t)grow * K + k0 + half * 16;
    __bf16* l = buf + r * LDA + half * 16;
    async_cp_b128(g, l);          // 8 elements
    async_cp_b128(g + 8, l + 8);  // 8 elements
  }
}

__device__ __forceinline__ void gemm_pair(const __bf16* __restrict__ A,
                                          const __bf16* __restrict__ Bp,
                                          int K, int M, int N,
                                          int m0, int n0w,
                                          int wm, int lane, int tid,
                                          __bf16* lds, v8f (&acc)[2][4]) {
  const int r    = tid >> 1;
  const int half = tid & 1;
  const int grow = m0 + r;
  __bf16* buf0 = lds;
  __bf16* buf1 = lds + BM * LDA;

  // zero out-of-range rows once (async loads are EXEC-masked for them)
  if (m0 + BM > M && grow >= M) {
    uint4 z = make_uint4(0, 0, 0, 0);
    uint4* s0 = reinterpret_cast<uint4*>(buf0 + r * LDA + half * 16);
    uint4* s1 = reinterpret_cast<uint4*>(buf1 + r * LDA + half * 16);
    s0[0] = z; s0[1] = z;
    s1[0] = z; s1[1] = z;
  }
  __syncthreads();

  issue_tile(A, K, M, grow, r, half, 0, buf0);   // slab 0 -> buf0
  issue_tile(A, K, M, grow, r, half, 32, buf1);  // slab 1 -> buf1

  int cur = 0;
  int k0  = 0;
  for (; k0 + 32 < K; k0 += 32) {
    WAIT_ASYNC(2);           // own loads for current slab landed (next stays in flight)
    __syncthreads();         // collective: buf[cur] complete
    compute_slab(cur ? buf1 : buf0, Bp, k0, N, n0w, wm, lane, acc);
    __syncthreads();         // all waves done reading buf[cur] (dscnt drained by wmma)
    issue_tile(A, K, M, grow, r, half, k0 + 64, cur ? buf1 : buf0);
    cur ^= 1;
  }
  WAIT_ASYNC(0);
  __syncthreads();
  compute_slab(cur ? buf1 : buf0, Bp, k0, N, n0w, wm, lane, acc);
  __syncthreads();
}

#else  // synchronous fallback (round-1 path)

__device__ __forceinline__ void gemm_pair(const __bf16* __restrict__ A,
                                          const __bf16* __restrict__ Bp,
                                          int K, int M, int N,
                                          int m0, int n0w,
                                          int wm, int lane, int tid,
                                          __bf16* lds, v8f (&acc)[2][4]) {
  const int r    = tid >> 1;
  const int half = tid & 1;
  const int grow = m0 + r;

  for (int k0 = 0; k0 < K; k0 += 32) {
    __syncthreads();
    uint4 q0 = make_uint4(0, 0, 0, 0), q1 = make_uint4(0, 0, 0, 0);
    if (grow < M) {
      const uint4* gp =
          reinterpret_cast<const uint4*>(A + (size_t)grow * K + k0 + half * 16);
      q0 = gp[0];
      q1 = gp[1];
    }
    uint4* sp = reinterpret_cast<uint4*>(lds + r * LDA + half * 16);
    sp[0] = q0;
    sp[1] = q1;
    __syncthreads();
    compute_slab(lds, Bp, k0, N, n0w, wm, lane, acc);
  }
  __syncthreads();
}

#endif

// ---------------------------------------------------------------------------
// fused (dual) GEMM: out = act(A1*B1 + [A2*B2] + bias1 + [bias2])
// ---------------------------------------------------------------------------
__global__ __launch_bounds__(256) void gemm_bf16_wmma(
    const __bf16* __restrict__ A1, const __bf16* __restrict__ B1, int K1,
    const __bf16* __restrict__ A2, const __bf16* __restrict__ B2, int K2,
    const float* __restrict__ bias1, const float* __restrict__ bias2,
    int M, int N, int do_relu,
    __bf16* __restrict__ out_bf, float* __restrict__ out_f32) {
#ifdef USE_ASYNC
  __shared__ __bf16 lds[2 * BM * LDA];
#else
  __shared__ __bf16 lds[BM * LDA];
#endif
  const int tid  = threadIdx.x;
  const int lane = tid & 31;
  const int wave = tid >> 5;
  const int wm   = wave & 3;             // 4 waves along M
  const int wn   = wave >> 2;            // 2 waves along N
  const int m0   = blockIdx.x * BM;
  const int n0w  = blockIdx.y * BN + wn * 64;

  v8f acc[2][4] = {};

  gemm_pair(A1, B1, K1, M, N, m0, n0w, wm, lane, tid, lds, acc);
  if (A2) gemm_pair(A2, B2, K2, M, N, m0, n0w, wm, lane, tid, lds, acc);

  // epilogue: C layout -> VGPR r: lanes 0-15 M=r, lanes 16-31 M=r+8; N = lane%16
#pragma unroll
  for (int i = 0; i < 2; ++i) {
#pragma unroll
    for (int j = 0; j < 4; ++j) {
      const int nb   = n0w + j * 16 + (lane & 15);
      float     badd = (bias1 ? bias1[nb] : 0.f) + (bias2 ? bias2[nb] : 0.f);
#pragma unroll
      for (int r = 0; r < 8; ++r) {
        const int m = m0 + wm * 32 + i * 16 + (lane >> 4) * 8 + r;
        if (m < M) {
          float v = acc[i][j][r] + badd;
          if (do_relu) v = fmaxf(v, 0.f);
          if (out_bf)
            out_bf[(size_t)m * N + nb] = (__bf16)v;
          else
            out_f32[(size_t)m * N + nb] = v;
        }
      }
    }
  }
}

// ---------------------------------------------------------------------------
// repack f32 [K x N] row-major weights into WMMA B-fragment order (bf16):
// frag (kblk, nblk) of 512 elems, within: idx = lane*16 + e,
// lane = (n%16) + 16*((k%32)/16), e = k%16
// ---------------------------------------------------------------------------
__global__ void pack_b_kernel(const float* __restrict__ w, __bf16* __restrict__ out,
                              int K, int N) {
  int idx = blockIdx.x * blockDim.x + threadIdx.x;
  if (idx >= K * N) return;
  int k = idx / N, n = idx % N;
  int kblk = k >> 5, nblk = n >> 4;
  int lane = (n & 15) + (((k >> 4) & 1) << 4);
  int e    = k & 15;
  size_t dst = ((size_t)kblk * (N >> 4) + nblk) * 512 + lane * 16 + e;
  out[dst] = (__bf16)w[idx];
}

__global__ void cvt_f32_bf16_kernel(const float* __restrict__ in,
                                    __bf16* __restrict__ out, int n) {
  int i = blockIdx.x * blockDim.x + threadIdx.x;
  if (i < n) out[i] = (__bf16)in[i];
}

// one edge per block, 256 threads * 2 channels = 512 channels
__global__ __launch_bounds__(256) void scatter_kernel(
    const __bf16* __restrict__ h, const float* __restrict__ ew,
    const int* __restrict__ edges, float* __restrict__ agg) {
  const int e   = blockIdx.x;
  const int t   = threadIdx.x;
  const int src = edges[e];
  const int dst = edges[NEDGE + e];
  const float w = ew[e];
  const __bf16* hp = h + (size_t)src * HID + t * 2;
  float*        ap = agg + (size_t)dst * HID + t * 2;
  unsafeAtomicAdd(ap,     (float)hp[0] * w);
  unsafeAtomicAdd(ap + 1, (float)hp[1] * w);
}

// ---------------------------------------------------------------------------
// driver
// ---------------------------------------------------------------------------
extern "C" void kernel_launch(void* const* d_in, const int* in_sizes, int n_in,
                              void* d_out, int out_size, void* d_ws, size_t ws_size,
                              hipStream_t stream) {
  (void)in_sizes; (void)n_in; (void)out_size; (void)ws_size;

  const float* x_drug     = (const float*)d_in[0];
  const float* x_dis      = (const float*)d_in[1];
  const float* ew0        = (const float*)d_in[2];
  const float* ew1        = (const float*)d_in[3];
  const int*   edge0      = (const int*)d_in[4];
  const int*   edge1      = (const int*)d_in[5];
  const float* lin_drug_w = (const float*)d_in[6];
  const float* lin_drug_b = (const float*)d_in[7];
  const float* lin_dis_w  = (const float*)d_in[8];
  const float* lin_dis_b  = (const float*)d_in[9];
  const float* rel_w      = (const float*)d_in[10];
  const float* rel_b      = (const float*)d_in[11];
  const float* root_w     = (const float*)d_in[12];
  const float* root_b     = (const float*)d_in[13];
  const float* fin_w      = (const float*)d_in[14];
  const float* fin_b      = (const float*)d_in[15];

  char*  ws  = (char*)d_ws;
  size_t off = 0;
  auto alloc = [&](size_t bytes) -> void* {
    void* p = ws + off;
    off += (bytes + 255) & ~(size_t)255;
    return p;
  };

  const size_t HN  = (size_t)NDRUG * HID;       // 25.6M elems
  float*  aggf     = (float*)alloc(HN * 4);     // f32 atomic accumulator (102.4MB)
  // x bf16 buffers alias aggf region (only used before first aggf memset)
  __bf16* xd_bf    = (__bf16*)aggf;
  __bf16* xs_bf    = (__bf16*)((char*)aggf + (size_t)NDRUG * INCH * 2);
  __bf16* hd_bf[2] = {(__bf16*)alloc(HN * 2), (__bf16*)alloc(HN * 2)};
  __bf16* hs_bf[2] = {(__bf16*)alloc(HN * 2), (__bf16*)alloc(HN * 2)};
  __bf16* aggs_bf  = (__bf16*)alloc(HN * 2);
  __bf16* aggd_bf  = (__bf16*)alloc(HN * 2);
  __bf16* pld      = (__bf16*)alloc((size_t)INCH * HID * 2);
  __bf16* pls      = (__bf16*)alloc((size_t)INCH * HID * 2);
  __bf16* prel     = (__bf16*)alloc((size_t)NLAYERS * 2 * HID * HID * 2);
  __bf16* proot    = (__bf16*)alloc((size_t)NLAYERS * 2 * HID * HID * 2);
  __bf16* pfin     = (__bf16*)alloc((size_t)HID * OUTCH * 2);

  const dim3 blk(256);

  // convert inputs to bf16
  {
    int n = NDRUG * INCH;
    cvt_f32_bf16_kernel<<<(n + 255) / 256, blk, 0, stream>>>(x_drug, xd_bf, n);
    cvt_f32_bf16_kernel<<<(n + 255) / 256, blk, 0, stream>>>(x_dis, xs_bf, n);
  }
  // repack all weights into WMMA fragment order
  pack_b_kernel<<<(INCH * HID + 255) / 256, blk, 0, stream>>>(lin_drug_w, pld, INCH, HID);
  pack_b_kernel<<<(INCH * HID + 255) / 256, blk, 0, stream>>>(lin_dis_w, pls, INCH, HID);
  for (int i = 0; i < NLAYERS * 2; ++i) {
    pack_b_kernel<<<(HID * HID + 255) / 256, blk, 0, stream>>>(
        rel_w + (size_t)i * HID * HID, prel + (size_t)i * HID * HID, HID, HID);
    pack_b_kernel<<<(HID * HID + 255) / 256, blk, 0, stream>>>(
        root_w + (size_t)i * HID * HID, proot + (size_t)i * HID * HID, HID, HID);
  }
  pack_b_kernel<<<(HID * OUTCH + 255) / 256, blk, 0, stream>>>(fin_w, pfin, HID, OUTCH);

  auto launch_gemm = [&](const __bf16* A1, const __bf16* B1, int K1,
                         const __bf16* A2, const __bf16* B2, int K2,
                         const float* b1, const float* b2, int M, int N, int relu,
                         __bf16* obf, float* of32) {
    dim3 grid((M + BM - 1) / BM, N / BN);
    gemm_bf16_wmma<<<grid, blk, 0, stream>>>(A1, B1, K1, A2, B2, K2, b1, b2, M, N,
                                             relu, obf, of32);
  };

  // input projections + ReLU
  launch_gemm(xd_bf, pld, INCH, nullptr, nullptr, 0, lin_drug_b, nullptr,
              NDRUG, HID, 1, hd_bf[0], nullptr);
  launch_gemm(xs_bf, pls, INCH, nullptr, nullptr, 0, lin_dis_b, nullptr,
              NDIS, HID, 1, hs_bf[0], nullptr);

  int cur = 0;
  for (int l = 0; l < NLAYERS; ++l) {
    int nxt = cur ^ 1;
    // edge type 0: drug -> disease
    (void)hipMemsetAsync(aggf, 0, HN * 4, stream);
    scatter_kernel<<<NEDGE, blk, 0, stream>>>(hd_bf[cur], ew0, edge0, aggf);
    cvt_f32_bf16_kernel<<<(int)((HN + 255) / 256), blk, 0, stream>>>(aggf, aggs_bf, (int)HN);
    // edge type 1: disease -> drug
    (void)hipMemsetAsync(aggf, 0, HN * 4, stream);
    scatter_kernel<<<NEDGE, blk, 0, stream>>>(hs_bf[cur], ew1, edge1, aggf);
    cvt_f32_bf16_kernel<<<(int)((HN + 255) / 256), blk, 0, stream>>>(aggf, aggd_bf, (int)HN);

    const __bf16* rel0  = prel  + (size_t)(l * 2 + 0) * HID * HID;
    const __bf16* rel1  = prel  + (size_t)(l * 2 + 1) * HID * HID;
    const __bf16* root0 = proot + (size_t)(l * 2 + 0) * HID * HID;
    const __bf16* root1 = proot + (size_t)(l * 2 + 1) * HID * HID;
    const float* rb0 = rel_b  + (size_t)(l * 2 + 0) * HID;
    const float* rb1 = rel_b  + (size_t)(l * 2 + 1) * HID;
    const float* ob0 = root_b + (size_t)(l * 2 + 0) * HID;
    const float* ob1 = root_b + (size_t)(l * 2 + 1) * HID;

    // out_s = relu(agg_s*rel0 + h_s*root0 + rb0 + ob0)
    launch_gemm(aggs_bf, rel0, HID, hs_bf[cur], root0, HID, rb0, ob0,
                NDIS, HID, 1, hs_bf[nxt], nullptr);
    // out_d = relu(agg_d*rel1 + h_d*root1 + rb1 + ob1)
    launch_gemm(aggd_bf, rel1, HID, hd_bf[cur], root1, HID, rb1, ob1,
                NDRUG, HID, 1, hd_bf[nxt], nullptr);
    cur = nxt;
  }

  // final shared projection -> f32 output [2, 50000, 256]
  float* outp = (float*)d_out;
  launch_gemm(hd_bf[cur], pfin, HID, nullptr, nullptr, 0, fin_b, nullptr,
              NDRUG, OUTCH, 0, nullptr, outp);
  launch_gemm(hs_bf[cur], pfin, HID, nullptr, nullptr, 0, fin_b, nullptr,
              NDIS, OUTCH, 0, nullptr, outp + (size_t)NDRUG * OUTCH);
}
